// CausalStreamTransformerBlock_33724083208866
// MI455X (gfx1250) — matrix operand order repaired and checked by reading, not compiled, not run on hardware
//
#include <hip/hip_runtime.h>
#include <cmath>

typedef __bf16 bf16;
typedef __attribute__((ext_vector_type(16))) __bf16 v16bf;
typedef __attribute__((ext_vector_type(8)))  float  v8f;

#define D_MODEL 512
#define NHEADS  8
#define HD      64
#define MAXC    4096
#define BB      8
#define TT      16
#define VVIS    256
#define SS      17          /* 1 cls + 16 tokens */
#define ROWS_Q  136         /* B*SS */
#define ROWS_QP 144         /* padded to 16 */
#define KVLEN   4368        /* 256 + 4096 + 16 */
#define NCHUNK  137         /* ceil(4368/32) */
#define NSPLIT  16
#define CPS     9           /* chunks per split (last split gets 2) */
#define LSTR    68          /* LDS row stride in floats (bank-conflict pad) */

// ---------------------------------------------------------------------------
// f32 -> bf16 elementwise convert (weights)
// ---------------------------------------------------------------------------
__global__ void cvt_f32_bf16(const float* __restrict__ in, bf16* __restrict__ out, int n) {
    int i = blockIdx.x * 256 + threadIdx.x;
    if (i < n) out[i] = (bf16)in[i];
}

__global__ void zero_bf16_rows(bf16* __restrict__ p, int startRow, int D) {
    int r = startRow + blockIdx.x;
    for (int c = threadIdx.x; c < D; c += blockDim.x) p[(size_t)r * D + c] = (bf16)0.0f;
}

// ---------------------------------------------------------------------------
// LayerNorm of one 512-wide row -> bf16 (zero-fills padding rows)
// ---------------------------------------------------------------------------
__global__ void ln_to_bf16(const float* __restrict__ in, const float* __restrict__ g,
                           const float* __restrict__ be, bf16* __restrict__ out, int validRows) {
    int r = blockIdx.x, tid = threadIdx.x;
    if (r >= validRows) {
        for (int c = tid; c < D_MODEL; c += 128) out[(size_t)r * D_MODEL + c] = (bf16)0.0f;
        return;
    }
    __shared__ float red[128];
    const float4 x = ((const float4*)(in + (size_t)r * D_MODEL))[tid];
    red[tid] = x.x + x.y + x.z + x.w;
    __syncthreads();
    for (int s = 64; s > 0; s >>= 1) { if (tid < s) red[tid] += red[tid + s]; __syncthreads(); }
    float mean = red[0] * (1.0f / D_MODEL);
    __syncthreads();
    float dx0 = x.x - mean, dx1 = x.y - mean, dx2 = x.z - mean, dx3 = x.w - mean;
    red[tid] = dx0 * dx0 + dx1 * dx1 + dx2 * dx2 + dx3 * dx3;
    __syncthreads();
    for (int s = 64; s > 0; s >>= 1) { if (tid < s) red[tid] += red[tid + s]; __syncthreads(); }
    float rs = rsqrtf(red[0] * (1.0f / D_MODEL) + 1e-5f);
    const float4 gv = ((const float4*)g)[tid];
    const float4 bv = ((const float4*)be)[tid];
    bf16* o = out + (size_t)r * D_MODEL + tid * 4;
    o[0] = (bf16)(dx0 * rs * gv.x + bv.x);
    o[1] = (bf16)(dx1 * rs * gv.y + bv.y);
    o[2] = (bf16)(dx2 * rs * gv.z + bv.z);
    o[3] = (bf16)(dx3 * rs * gv.w + bv.w);
}

// ---------------------------------------------------------------------------
// Build combined (cls + tokens) f32, LN -> qn bf16, and cur rows -> curn bf16
// ---------------------------------------------------------------------------
__global__ void build_combined(const float* __restrict__ prevcls, const float* __restrict__ curtok,
                               const float* __restrict__ g, const float* __restrict__ be,
                               float* __restrict__ combined, bf16* __restrict__ qn,
                               bf16* __restrict__ curn) {
    int r = blockIdx.x, tid = threadIdx.x;
    if (r >= ROWS_Q) {
        for (int c = tid; c < D_MODEL; c += 128) {
            combined[(size_t)r * D_MODEL + c] = 0.0f;
            qn[(size_t)r * D_MODEL + c] = (bf16)0.0f;
        }
        return;
    }
    int b = r / SS, s = r % SS;
    const float* src = (s == 0) ? (prevcls + (size_t)b * D_MODEL)
                                : (curtok + ((size_t)b * TT + (s - 1)) * D_MODEL);
    const float4 x = ((const float4*)src)[tid];
    ((float4*)(combined + (size_t)r * D_MODEL))[tid] = x;

    __shared__ float red[128];
    red[tid] = x.x + x.y + x.z + x.w;
    __syncthreads();
    for (int st = 64; st > 0; st >>= 1) { if (tid < st) red[tid] += red[tid + st]; __syncthreads(); }
    float mean = red[0] * (1.0f / D_MODEL);
    __syncthreads();
    float dx0 = x.x - mean, dx1 = x.y - mean, dx2 = x.z - mean, dx3 = x.w - mean;
    red[tid] = dx0 * dx0 + dx1 * dx1 + dx2 * dx2 + dx3 * dx3;
    __syncthreads();
    for (int st = 64; st > 0; st >>= 1) { if (tid < st) red[tid] += red[tid + st]; __syncthreads(); }
    float rs = rsqrtf(red[0] * (1.0f / D_MODEL) + 1e-5f);
    const float4 gv = ((const float4*)g)[tid];
    const float4 bv = ((const float4*)be)[tid];
    float y0 = dx0 * rs * gv.x + bv.x, y1 = dx1 * rs * gv.y + bv.y;
    float y2 = dx2 * rs * gv.z + bv.z, y3 = dx3 * rs * gv.w + bv.w;
    bf16* q = qn + (size_t)r * D_MODEL + tid * 4;
    q[0] = (bf16)y0; q[1] = (bf16)y1; q[2] = (bf16)y2; q[3] = (bf16)y3;
    if (s > 0) {
        bf16* cu = curn + ((size_t)b * TT + (s - 1)) * D_MODEL + tid * 4;
        cu[0] = (bf16)y0; cu[1] = (bf16)y1; cu[2] = (bf16)y2; cu[3] = (bf16)y3;
    }
}

// ---------------------------------------------------------------------------
// Generic bf16 WMMA GEMM: out = act(A[MxK] @ W[KxN] + bias [+ resid])
// 1 wave per 16x16 tile. M,N multiples of 16; K multiple of 32.
// ---------------------------------------------------------------------------
__global__ void gemm_bf16(const bf16* __restrict__ A, const bf16* __restrict__ W,
                          const float* __restrict__ bias, const float* __restrict__ resid,
                          float* __restrict__ outF, bf16* __restrict__ outB,
                          int M, int N, int K, int gelu) {
    const int lane = threadIdx.x & 31;
    const int ln = lane & 15, hi = lane >> 4;
    const int m0 = blockIdx.x * 16, n0 = blockIdx.y * 16;

    v8f acc;
    for (int i = 0; i < 8; ++i) acc[i] = 0.0f;

    const bf16* arow = A + (size_t)(m0 + ln) * K + (hi << 3);
    union Frag { uint4 q[2]; v16bf v; };

    for (int k0 = 0; k0 < K; k0 += 32) {
        Frag a, b;
        a.q[0] = *(const uint4*)(arow + k0);
        a.q[1] = *(const uint4*)(arow + k0 + 16);
        const bf16* bp = W + (size_t)(k0 + (hi << 3)) * N + n0 + ln;
        for (int j = 0; j < 16; ++j)
            b.v[j] = bp[(size_t)((j < 8) ? j : j + 8) * N];
        if (k0 + 32 < K) {
            __builtin_prefetch((const void*)(arow + k0 + 32), 0, 1);
            __builtin_prefetch((const void*)(bp + (size_t)32 * N), 0, 1);
        }
        acc = __builtin_amdgcn_wmma_f32_16x16x32_bf16(false, a.v, false, b.v,
                                                      (short)0, acc, false, false);
    }

    const int n = n0 + ln;
    const float bsv = bias[n];
    for (int r = 0; r < 8; ++r) {
        int m = m0 + r + (hi << 3);
        float v = acc[r] + bsv;
        if (resid) v += resid[(size_t)m * N + n];
        if (gelu)  v = 0.5f * v * (1.0f + erff(v * 0.70710678118f));
        if (outF) outF[(size_t)m * N + n] = v;
        if (outB) outB[(size_t)m * N + n] = (bf16)v;
    }
}

// ---------------------------------------------------------------------------
// Split-K flash attention partial. Grid (B*H, NSPLIT), 1 wave per block.
// KV chunks (32 keys x 64 dims, f32) are DMA'd into LDS with
// GLOBAL_LOAD_ASYNC_TO_LDS_B128 (double buffered, ASYNCcnt-tracked), then
// converted to bf16 WMMA fragments from LDS. Online softmax per split;
// partial (m, l, unnormalized acc) written to workspace.
// ---------------------------------------------------------------------------
__global__ void flash_partial(const float* __restrict__ qf,
                              const float* __restrict__ kvis, const float* __restrict__ vvis,
                              const float* __restrict__ kcache, const float* __restrict__ vcache,
                              const float* __restrict__ kcur, const float* __restrict__ vcur,
                              const int* __restrict__ tmask, const int* __restrict__ clenArr,
                              float* __restrict__ pm, float* __restrict__ pl,
                              float* __restrict__ pacc) {
    const int bh = blockIdx.x;
    const int sp = blockIdx.y;
    const int b = bh >> 3, h = bh & 7;
    const int lane = threadIdx.x;
    const int ln = lane & 15, hi = lane >> 4;
    const int hi8 = hi << 3;
    const int clen = clenArr[b];

    const int cbase = sp * CPS;
    const int cnt = (NCHUNK - cbase) < CPS ? (NCHUNK - cbase) : CPS;

    __shared__ float Kl[2][32 * LSTR];
    __shared__ float Vl[2][32 * LSTR];
    __shared__ float S[2][16][32];
    __shared__ float mrow[2][16], lrow[2][16], frow[2][16];

    mrow[hi][ln] = -1e30f;
    lrow[hi][ln] = 0.0f;
    __syncthreads();

    // Q fragments (softmax scale folded in); zero for padding rows >= 17
    v16bf qa[2][2];
    for (int i = 0; i < 2; ++i) {
        int m = i * 16 + ln;
        for (int ks = 0; ks < 2; ++ks) {
            for (int j = 0; j < 16; ++j) {
                int koff = ((j < 8) ? j : j + 8) + hi8;
                float v = 0.0f;
                if (m < SS)
                    v = qf[(size_t)(b * SS + m) * D_MODEL + h * HD + ks * 32 + koff] * 0.125f;
                qa[i][ks][j] = (bf16)v;
            }
        }
    }

    v8f acc[2][4];
    for (int i = 0; i < 2; ++i)
        for (int nt = 0; nt < 4; ++nt)
            for (int r = 0; r < 8; ++r) acc[i][nt][r] = 0.0f;

    // segment resolver for a chunk starting at key index c0
    auto segment = [&](int c0, const float*& Kb, const float*& Vb, int& rstr, int& rows) {
        if (c0 < VVIS) {
            Kb = kvis + ((size_t)(b * VVIS + c0)) * D_MODEL + h * HD;
            Vb = vvis + ((size_t)(b * VVIS + c0)) * D_MODEL + h * HD;
            rstr = D_MODEL; rows = 32;
        } else if (c0 < VVIS + MAXC) {
            int r0 = c0 - VVIS;
            Kb = kcache + (((size_t)(b * NHEADS + h)) * MAXC + r0) * HD;
            Vb = vcache + (((size_t)(b * NHEADS + h)) * MAXC + r0) * HD;
            rstr = HD; rows = 32;
        } else {
            Kb = kcur + ((size_t)(b * TT)) * D_MODEL + h * HD;
            Vb = vcur + ((size_t)(b * TT)) * D_MODEL + h * HD;
            rstr = D_MODEL; rows = 16;
        }
    };

    const unsigned kbase0 = (unsigned)(size_t)&Kl[0][0];
    const unsigned kbase1 = (unsigned)(size_t)&Kl[1][0];
    const unsigned vbase0 = (unsigned)(size_t)&Vl[0][0];
    const unsigned vbase1 = (unsigned)(size_t)&Vl[1][0];

    // issue async DMA of one 32x64 f32 K/V chunk into LDS buffer `buf`
    auto issue_chunk = [&](int ci, int buf) {
        const float *Kb, *Vb;
        int rstr, rows;
        int c0 = (cbase + ci) * 32;
        segment(c0, Kb, Vb, rstr, rows);
        unsigned kb = buf ? kbase1 : kbase0;
        unsigned vb = buf ? vbase1 : vbase0;
#pragma unroll
        for (int it = 0; it < 16; ++it) {
            int e = it * 32 + lane;           // 512 x b128 per chunk pair
            int row = e >> 4;                 // 0..31
            int q4 = (e & 15) << 2;           // float offset within row
            int kr = row < rows ? row : rows - 1;
            unsigned lak = kb + (unsigned)((row * LSTR + q4) * 4);
            unsigned lav = vb + (unsigned)((row * LSTR + q4) * 4);
            unsigned long long gak = (unsigned long long)(size_t)(Kb + (size_t)kr * rstr + q4);
            unsigned long long gav = (unsigned long long)(size_t)(Vb + (size_t)kr * rstr + q4);
            asm volatile("global_load_async_to_lds_b128 %0, %1, off"
                         :: "v"(lak), "v"(gak) : "memory");
            asm volatile("global_load_async_to_lds_b128 %0, %1, off"
                         :: "v"(lav), "v"(gav) : "memory");
        }
    };

    issue_chunk(0, 0);

    for (int ci = 0; ci < cnt; ++ci) {
        const int buf = ci & 1;
        const float* Ksh = Kl[buf];
        const float* Vsh = Vl[buf];
        int c0 = (cbase + ci) * 32;

        if (ci + 1 < cnt) {
            issue_chunk(ci + 1, buf ^ 1);
            asm volatile("s_wait_asynccnt 0x20" ::: "memory");  // current chunk done
        } else {
            asm volatile("s_wait_asynccnt 0x0" ::: "memory");
        }
        __syncthreads();

        // scores: S[i][t] = Q[i] . K^T  (head dim 64 = two 32-wide WMMA steps)
        v8f sc[2][2];
        for (int i = 0; i < 2; ++i)
            for (int t = 0; t < 2; ++t)
                for (int r = 0; r < 8; ++r) sc[i][t][r] = 0.0f;

        for (int t = 0; t < 2; ++t) {
            for (int ks = 0; ks < 2; ++ks) {
                v16bf kfr;
                for (int j = 0; j < 16; ++j)
                    kfr[j] = (bf16)Ksh[(t * 16 + ln) * LSTR + ks * 32 + ((j < 8) ? j : j + 8) + hi8];
                for (int i = 0; i < 2; ++i)
                    sc[i][t] = __builtin_amdgcn_wmma_f32_16x16x32_bf16(
                        false, qa[i][ks], false, kfr, (short)0, sc[i][t], false, false);
            }
        }

        // masking (score column == lane%16 across all accumulator rows)
        for (int t = 0; t < 2; ++t) {
            int kg = c0 + t * 16 + ln;
            bool valid;
            if (kg < VVIS) valid = true;
            else if (kg < VVIS + MAXC) valid = (kg - VVIS) < clen;
            else if (kg < KVLEN) valid = tmask[b * TT + (kg - (VVIS + MAXC))] != 0;
            else valid = false;
            if (!valid)
                for (int i = 0; i < 2; ++i)
                    for (int r = 0; r < 8; ++r) sc[i][t][r] = -1e30f;
        }

        // stage scores in LDS (C layout -> row major)
        for (int i = 0; i < 2; ++i)
            for (int t = 0; t < 2; ++t)
                for (int r = 0; r < 8; ++r)
                    S[i][r + hi8][t * 16 + ln] = sc[i][t][r];
        __syncthreads();

        // per-row online softmax stats: lane -> (tile hi, row ln)
        {
            float mold = mrow[hi][ln];
            float mx = mold;
            for (int n = 0; n < 32; ++n) mx = fmaxf(mx, S[hi][ln][n]);
            float f = __expf(mold - mx);
            float ssum = 0.0f;
            for (int n = 0; n < 32; ++n) ssum += __expf(S[hi][ln][n] - mx);
            lrow[hi][ln] = lrow[hi][ln] * f + ssum;
            mrow[hi][ln] = mx;
            frow[hi][ln] = f;
        }
        __syncthreads();

        // rescale ctx accumulators
        for (int i = 0; i < 2; ++i)
            for (int r = 0; r < 8; ++r) {
                float f = frow[i][r + hi8];
                for (int nt = 0; nt < 4; ++nt) acc[i][nt][r] *= f;
            }

        // probability A-fragments from LDS
        v16bf p[2];
        for (int i = 0; i < 2; ++i) {
            float mn = mrow[i][ln];
            for (int j = 0; j < 16; ++j) {
                int k = ((j < 8) ? j : j + 8) + hi8;
                p[i][j] = (bf16)__expf(S[i][ln][k] - mn);
            }
        }

        // ctx += P @ V  (K-dim = 32 keys; 4 output tiles of 16 dims)
        for (int nt = 0; nt < 4; ++nt) {
            v16bf vfr;
            for (int j = 0; j < 16; ++j)
                vfr[j] = (bf16)Vsh[(((j < 8) ? j : j + 8) + hi8) * LSTR + nt * 16 + ln];
            for (int i = 0; i < 2; ++i)
                acc[i][nt] = __builtin_amdgcn_wmma_f32_16x16x32_bf16(
                    false, p[i], false, vfr, (short)0, acc[i][nt], false, false);
        }
        __syncthreads();
    }

    // write partial results (unnormalized)
    const size_t pb = ((size_t)bh * NSPLIT + sp) * 32;
    pm[pb + hi * 16 + ln] = mrow[hi][ln];
    pl[pb + hi * 16 + ln] = lrow[hi][ln];
    for (int i = 0; i < 2; ++i)
        for (int r = 0; r < 8; ++r) {
            int row = i * 16 + r + hi8;
            for (int nt = 0; nt < 4; ++nt)
                pacc[(pb + row) * HD + nt * 16 + ln] = acc[i][nt][r];
        }
}

// ---------------------------------------------------------------------------
// combine split partials -> ctx (bf16). Grid = B*H, block = 64 (d = tid).
// ---------------------------------------------------------------------------
__global__ void flash_combine(const float* __restrict__ pm, const float* __restrict__ pl,
                              const float* __restrict__ pacc, bf16* __restrict__ ctxb) {
    const int bh = blockIdx.x;
    const int b = bh >> 3, h = bh & 7;
    const int d = threadIdx.x;
    for (int row = 0; row < SS; ++row) {
        float M = -1e30f;
        for (int s = 0; s < NSPLIT; ++s)
            M = fmaxf(M, pm[((size_t)bh * NSPLIT + s) * 32 + row]);
        float L = 0.0f, sum = 0.0f;
        for (int s = 0; s < NSPLIT; ++s) {
            size_t pb = ((size_t)bh * NSPLIT + s) * 32;
            float w = __expf(pm[pb + row] - M);
            L += pl[pb + row] * w;
            sum += pacc[(pb + row) * HD + d] * w;
        }
        float v = (L > 0.0f) ? sum / L : 0.0f;
        ctxb[(size_t)(b * SS + row) * D_MODEL + h * HD + d] = (bf16)v;
    }
}

// ---------------------------------------------------------------------------
// next_cls / next_tokens (+ ntok workspace copy for the cache-LN)
// ---------------------------------------------------------------------------
__global__ void finalize_k(const float* __restrict__ c3, const float* __restrict__ prevcls,
                           const int* __restrict__ tmask, const int* __restrict__ act,
                           float* __restrict__ ocls, float* __restrict__ otok,
                           float* __restrict__ ntok) {
    int r = blockIdx.x, tid = threadIdx.x;
    int b = r / SS, s = r % SS;
    const float* row = c3 + (size_t)r * D_MODEL;
    if (s == 0) {
        const float* src = act[b] ? row : (prevcls + (size_t)b * D_MODEL);
        for (int c = tid; c < D_MODEL; c += 128) ocls[(size_t)b * D_MODEL + c] = src[c];
    } else {
        int t = s - 1;
        float mk = (tmask[b * TT + t] != 0) ? 1.0f : 0.0f;
        for (int c = tid; c < D_MODEL; c += 128) {
            float v = row[c] * mk;
            otok[((size_t)b * TT + t) * D_MODEL + c] = v;
            ntok[((size_t)b * TT + t) * D_MODEL + c] = v;
        }
    }
}

// ---------------------------------------------------------------------------
// scatter-append new K/V rows into the copied caches
// ---------------------------------------------------------------------------
__global__ void append_kv(const float* __restrict__ kapp, const float* __restrict__ vapp,
                          const int* __restrict__ tmask, const int* __restrict__ act,
                          const int* __restrict__ clen,
                          float* __restrict__ okey, float* __restrict__ oval) {
    int bt = blockIdx.x;
    int b = bt >> 4, t = bt & 15;
    if (!tmask[b * TT + t] || !act[b]) return;
    int c = 0;
    for (int j = 0; j <= t; ++j) c += (tmask[b * TT + j] != 0);
    int dest = clen[b] + c - 1;
    if (dest < 0 || dest >= MAXC) return;
    for (int col = threadIdx.x; col < D_MODEL; col += blockDim.x) {
        int h = col >> 6, d = col & 63;
        size_t o = (((size_t)b * NHEADS + h) * MAXC + dest) * HD + d;
        okey[o] = kapp[((size_t)b * TT + t) * D_MODEL + col];
        oval[o] = vapp[((size_t)b * TT + t) * D_MODEL + col];
    }
}

__global__ void write_len_k(const int* __restrict__ tmask, const int* __restrict__ act,
                            const int* __restrict__ clen, float* __restrict__ olen) {
    int b = threadIdx.x;
    if (b >= BB) return;
    int s = 0;
    for (int j = 0; j < TT; ++j) s += (tmask[b * TT + j] != 0);
    olen[b] = (float)(act[b] ? clen[b] + s : clen[b]);
}

// ---------------------------------------------------------------------------
extern "C" void kernel_launch(void* const* d_in, const int* in_sizes, int n_in,
                              void* d_out, int out_size, void* d_ws, size_t ws_size,
                              hipStream_t stream) {
    const float* prevcls = (const float*)d_in[0];
    const float* curtok  = (const float*)d_in[1];
    const float* vistok  = (const float*)d_in[2];
    const float* ckey    = (const float*)d_in[3];
    const float* cval    = (const float*)d_in[4];
    const float* q_w = (const float*)d_in[5];  const float* q_b = (const float*)d_in[6];
    const float* k_w = (const float*)d_in[7];  const float* k_b = (const float*)d_in[8];
    const float* v_w = (const float*)d_in[9];  const float* v_b = (const float*)d_in[10];
    const float* o_w = (const float*)d_in[11]; const float* o_b = (const float*)d_in[12];
    const float* f1_w = (const float*)d_in[13]; const float* f1_b = (const float*)d_in[14];
    const float* f2_w = (const float*)d_in[15]; const float* f2_b = (const float*)d_in[16];
    const float* qn_g = (const float*)d_in[17]; const float* qn_b = (const float*)d_in[18];
    const float* vn_g = (const float*)d_in[19]; const float* vn_b = (const float*)d_in[20];
    const float* fn_g = (const float*)d_in[21]; const float* fn_b = (const float*)d_in[22];
    const float* cn_g = (const float*)d_in[23]; const float* cn_b = (const float*)d_in[24];
    const int* tmask = (const int*)d_in[25];
    const int* act   = (const int*)d_in[26];
    const int* clen  = (const int*)d_in[27];

    float* out   = (float*)d_out;
    float* o_cls = out;                                    // 8*512
    float* o_tok = o_cls + BB * D_MODEL;                   // 8*16*512
    float* o_key = o_tok + BB * TT * D_MODEL;              // 8*8*4096*64
    float* o_val = o_key + (size_t)BB * NHEADS * MAXC * HD;
    float* o_len = o_val + (size_t)BB * NHEADS * MAXC * HD;

    char* wsb = (char*)d_ws;
    size_t off = 0;
    auto alloc = [&](size_t bytes) -> char* {
        char* p = wsb + off;
        off = (off + bytes + 255) & ~(size_t)255;
        return p;
    };
    bf16* wq  = (bf16*)alloc(512 * 512 * 2);
    bf16* wk  = (bf16*)alloc(512 * 512 * 2);
    bf16* wv  = (bf16*)alloc(512 * 512 * 2);
    bf16* wo  = (bf16*)alloc(512 * 512 * 2);
    bf16* wf1 = (bf16*)alloc(512 * 2048 * 2);
    bf16* wf2 = (bf16*)alloc(2048 * 512 * 2);
    float* combined = (float*)alloc(ROWS_QP * 512 * 4);
    bf16*  qn    = (bf16*)alloc(ROWS_QP * 512 * 2);
    bf16*  curn  = (bf16*)alloc(128 * 512 * 2);
    bf16*  visn  = (bf16*)alloc(2048 * 512 * 2);
    float* qf    = (float*)alloc(ROWS_QP * 512 * 4);
    float* kvis  = (float*)alloc(2048 * 512 * 4);
    float* vvis  = (float*)alloc(2048 * 512 * 4);
    float* kcur  = (float*)alloc(128 * 512 * 4);
    float* vcur  = (float*)alloc(128 * 512 * 4);
    bf16*  ctxb  = (bf16*)alloc(ROWS_QP * 512 * 2);
    float* comb2 = (float*)alloc(ROWS_QP * 512 * 4);
    bf16*  fnb   = (bf16*)alloc(ROWS_QP * 512 * 2);
    bf16*  hb    = (bf16*)alloc(ROWS_QP * 2048 * 2);
    float* comb3 = (float*)alloc(ROWS_QP * 512 * 4);
    float* ntok  = (float*)alloc(128 * 512 * 4);
    bf16*  cnb   = (bf16*)alloc(128 * 512 * 2);
    float* kapp  = (float*)alloc(128 * 512 * 4);
    float* vapp  = (float*)alloc(128 * 512 * 4);
    float* pm    = (float*)alloc((size_t)64 * NSPLIT * 32 * 4);
    float* pl    = (float*)alloc((size_t)64 * NSPLIT * 32 * 4);
    float* pacc  = (float*)alloc((size_t)64 * NSPLIT * 32 * HD * 4);

    // bulk cache copy into outputs (front of the stream)
    size_t cacheBytes = (size_t)BB * NHEADS * MAXC * HD * sizeof(float);
    hipMemcpyAsync(o_key, ckey, cacheBytes, hipMemcpyDeviceToDevice, stream);
    hipMemcpyAsync(o_val, cval, cacheBytes, hipMemcpyDeviceToDevice, stream);

    // weights -> bf16
    cvt_f32_bf16<<<1024, 256, 0, stream>>>(q_w, wq, 512 * 512);
    cvt_f32_bf16<<<1024, 256, 0, stream>>>(k_w, wk, 512 * 512);
    cvt_f32_bf16<<<1024, 256, 0, stream>>>(v_w, wv, 512 * 512);
    cvt_f32_bf16<<<1024, 256, 0, stream>>>(o_w, wo, 512 * 512);
    cvt_f32_bf16<<<4096, 256, 0, stream>>>(f1_w, wf1, 512 * 2048);
    cvt_f32_bf16<<<4096, 256, 0, stream>>>(f2_w, wf2, 2048 * 512);

    // LayerNorms
    build_combined<<<ROWS_QP, 128, 0, stream>>>(prevcls, curtok, qn_g, qn_b,
                                                combined, qn, curn);
    ln_to_bf16<<<2048, 128, 0, stream>>>(vistok, vn_g, vn_b, visn, 2048);

    // projections (bf16 WMMA)
    gemm_bf16<<<dim3(ROWS_QP / 16, 32), 32, 0, stream>>>(qn, wq, q_b, nullptr, qf, nullptr,
                                                         ROWS_QP, 512, 512, 0);
    gemm_bf16<<<dim3(128, 32), 32, 0, stream>>>(visn, wk, k_b, nullptr, kvis, nullptr,
                                                2048, 512, 512, 0);
    gemm_bf16<<<dim3(128, 32), 32, 0, stream>>>(visn, wv, v_b, nullptr, vvis, nullptr,
                                                2048, 512, 512, 0);
    gemm_bf16<<<dim3(8, 32), 32, 0, stream>>>(curn, wk, k_b, nullptr, kcur, nullptr,
                                              128, 512, 512, 0);
    gemm_bf16<<<dim3(8, 32), 32, 0, stream>>>(curn, wv, v_b, nullptr, vcur, nullptr,
                                              128, 512, 512, 0);

    // attention: split-K partials (async-LDS staged) + combine
    zero_bf16_rows<<<ROWS_QP - ROWS_Q, 128, 0, stream>>>(ctxb, ROWS_Q, 512);
    flash_partial<<<dim3(BB * NHEADS, NSPLIT), 32, 0, stream>>>(
        qf, kvis, vvis, ckey, cval, kcur, vcur, tmask, clen, pm, pl, pacc);
    flash_combine<<<BB * NHEADS, HD, 0, stream>>>(pm, pl, pacc, ctxb);

    // output projection + residual
    gemm_bf16<<<dim3(ROWS_QP / 16, 32), 32, 0, stream>>>(ctxb, wo, o_b, combined,
                                                         comb2, nullptr, ROWS_QP, 512, 512, 0);
    // FFN
    ln_to_bf16<<<ROWS_QP, 128, 0, stream>>>(comb2, fn_g, fn_b, fnb, ROWS_Q);
    gemm_bf16<<<dim3(ROWS_QP / 16, 128), 32, 0, stream>>>(fnb, wf1, f1_b, nullptr,
                                                          nullptr, hb, ROWS_QP, 2048, 512, 1);
    gemm_bf16<<<dim3(ROWS_QP / 16, 32), 32, 0, stream>>>(hb, wf2, f2_b, comb2,
                                                         comb3, nullptr, ROWS_QP, 512, 2048, 0);

    // outputs + cache append
    finalize_k<<<ROWS_Q, 128, 0, stream>>>(comb3, prevcls, tmask, act, o_cls, o_tok, ntok);
    ln_to_bf16<<<128, 128, 0, stream>>>(ntok, cn_g, cn_b, cnb, 128);
    gemm_bf16<<<dim3(8, 32), 32, 0, stream>>>(cnb, wk, k_b, nullptr, kapp, nullptr,
                                              128, 512, 512, 0);
    gemm_bf16<<<dim3(8, 32), 32, 0, stream>>>(cnb, wv, v_b, nullptr, vapp, nullptr,
                                              128, 512, 512, 0);
    append_kv<<<BB * TT, 128, 0, stream>>>(kapp, vapp, tmask, act, clen, o_key, o_val);
    write_len_k<<<1, 32, 0, stream>>>(tmask, act, clen, o_len);
}
